// SNNModel_19834158973435
// MI455X (gfx1250) — matrix operand order repaired and checked by reading, not compiled, not run on hardware
//
#include <hip/hip_runtime.h>
#include <hip/hip_bf16.h>

#define BATCH 256
#define TSEQ  64
#define HID   512
#define GATES 2048   // 4*HID
#define NSPK  100

typedef __attribute__((ext_vector_type(16))) __bf16 v16bf;
typedef __attribute__((ext_vector_type(8)))  float  v8f;

// ---------------------------------------------------------------------------
// WMMA fragment helpers (CDNA5 wave32 layouts, ISA 7.12.2)
// ---------------------------------------------------------------------------

// A: 16x32 bf16 tile, rows m0..m0+15, K window [k0,k0+32).
// lane = half*16 + (m-m0); per lane: elems 0..7 = K kb..kb+7, elems 8..15 = K kb+16..kb+23
// where kb = k0 + half*8.  row_stride lets us stride through x[b][t][:] by batch.
__device__ __forceinline__ v16bf load_a_frag(const __bf16* __restrict__ base,
                                             int row_stride, int m0, int k0, int lane) {
  int m  = m0 + (lane & 15);
  int kb = k0 + ((lane >> 4) << 3);
  const __bf16* p = base + (size_t)m * row_stride + kb;
  v16bf a;
#pragma unroll
  for (int v = 0; v < 8; ++v) {
    int off = (v < 4) ? 2 * v : 16 + 2 * (v - 4);
    a[2 * v]     = p[off];
    a[2 * v + 1] = p[off + 1];
  }
  return a;
}

// B: 32x16 bf16 tile where B[k][n] = W[n0+n][k]  (W row-major [N,K]).
// lane = half*16 + n; per lane: elems 0..7 = K kb..kb+7, elems 8..15 = K kb+16..kb+23, kb = k0 + half*8.
__device__ __forceinline__ v16bf load_b_frag(const __bf16* __restrict__ W,
                                             int row_stride, int n0, int k0, int lane) {
  int n  = n0 + (lane & 15);
  int kb = k0 + ((lane >> 4) << 3);
  const __bf16* p = W + (size_t)n * row_stride + kb;
  v16bf b;
#pragma unroll
  for (int e = 0; e < 16; ++e) b[e] = p[(e < 8) ? e : e + 8];
  return b;
}

// A-fragment built by thresholding fp32 mem on the fly: spike = (mem > 0.5)
__device__ __forceinline__ v16bf load_spike_a(const float* __restrict__ mem,
                                              int m0, int k0, int lane) {
  int m  = m0 + (lane & 15);
  int kb = k0 + ((lane >> 4) << 3);
  const float* p = mem + (size_t)m * HID + kb;
  v16bf a;
#pragma unroll
  for (int v = 0; v < 8; ++v) {
    int off = (v < 4) ? 2 * v : 16 + 2 * (v - 4);
    a[2 * v]     = (__bf16)((p[off]     > 0.5f) ? 1.0f : 0.0f);
    a[2 * v + 1] = (__bf16)((p[off + 1] > 0.5f) ? 1.0f : 0.0f);
  }
  return a;
}

__device__ __forceinline__ v8f wmma_bf16(v16bf a, v16bf b, v8f c) {
  return __builtin_amdgcn_wmma_f32_16x16x32_bf16(false, a, false, b, (short)0, c,
                                                 false, false);
}

__device__ __forceinline__ float sigmoidf_(float x) {
  return 1.0f / (1.0f + __expf(-x));
}

// ---------------------------------------------------------------------------
// Generic C = A(bf16) @ W(bf16)^T + bias, one 16x16 tile per wave.
// Used for G = x @ W_ih^T + (b_ih+b_hh)  and  mem0 = h_last @ W_fc^T + b_fc.
// ---------------------------------------------------------------------------
__global__ __launch_bounds__(256)
void gemm_bias_kernel(const __bf16* __restrict__ A, int lda,
                      const __bf16* __restrict__ W, int ldw,
                      const float* __restrict__ bias,
                      float* __restrict__ C, int ldc,
                      int ntn, int K) {
  int lane = threadIdx.x;
  int tile = blockIdx.x * blockDim.y + threadIdx.y;
  int mt = tile / ntn, nt = tile % ntn;
  int m0 = mt * 16, n0 = nt * 16;
  int nc = n0 + (lane & 15);
  float bv = bias[nc];
  v8f acc;
#pragma unroll
  for (int v = 0; v < 8; ++v) acc[v] = bv;
  for (int k0 = 0; k0 < K; k0 += 32) {
    v16bf a = load_a_frag(A, lda, m0, k0, lane);
    v16bf b = load_b_frag(W, ldw, n0, k0, lane);
    acc = wmma_bf16(a, b, acc);
  }
  int mrow = m0 + ((lane >> 4) << 3);
#pragma unroll
  for (int v = 0; v < 8; ++v) C[(size_t)(mrow + v) * ldc + nc] = acc[v];
}

// ---------------------------------------------------------------------------
// One LSTM timestep: gates = G[:,t,:] (or bias + x_t@W_ih^T) + h@W_hh^T, then
// i,f,g,o nonlinearity, c/h update.  One 16(batch) x 16(hid) tile per wave;
// each wave carries all 4 gate accumulators so the A-fragment is reused 4x.
// ---------------------------------------------------------------------------
__global__ __launch_bounds__(256)
void lstm_step_kernel(const float* __restrict__ G,     // [B*T, 4H] or nullptr
                      const __bf16* __restrict__ xb,   // [B, T, I] bf16
                      const __bf16* __restrict__ Wih,  // [4H, I] bf16
                      const __bf16* __restrict__ Whh,  // [4H, H] bf16
                      const float* __restrict__ bias,  // [4H] = b_ih + b_hh
                      const __bf16* __restrict__ hprev,// [B, H] bf16
                      __bf16* __restrict__ hnext,      // [B, H] bf16
                      float* __restrict__ c,           // [B, H] f32 (in place)
                      int t) {
  int lane = threadIdx.x;
  int tile = blockIdx.x * blockDim.y + threadIdx.y;   // 16 x 32 = 512 tiles
  int mt = tile / (HID / 16), nt = tile % (HID / 16);
  int m0 = mt * 16, h0 = nt * 16;
  int nc = lane & 15;
  int half8 = (lane >> 4) << 3;

  v8f acc[4];
  if (G != nullptr) {
#pragma unroll
    for (int q = 0; q < 4; ++q)
#pragma unroll
      for (int v = 0; v < 8; ++v)
        acc[q][v] = G[(size_t)((m0 + half8 + v) * TSEQ + t) * GATES
                      + q * HID + h0 + nc];
  } else {
#pragma unroll
    for (int q = 0; q < 4; ++q) {
      float bv = bias[q * HID + h0 + nc];
#pragma unroll
      for (int v = 0; v < 8; ++v) acc[q][v] = bv;
    }
    const __bf16* xbase = xb + (size_t)t * HID;       // batch row stride T*I
    for (int k0 = 0; k0 < HID; k0 += 32) {
      v16bf a = load_a_frag(xbase, TSEQ * HID, m0, k0, lane);
#pragma unroll
      for (int q = 0; q < 4; ++q) {
        v16bf b = load_b_frag(Wih, HID, q * HID + h0, k0, lane);
        acc[q] = wmma_bf16(a, b, acc[q]);
      }
    }
  }

  for (int k0 = 0; k0 < HID; k0 += 32) {
    v16bf a = load_a_frag(hprev, HID, m0, k0, lane);
#pragma unroll
    for (int q = 0; q < 4; ++q) {
      v16bf b = load_b_frag(Whh, HID, q * HID + h0, k0, lane);
      acc[q] = wmma_bf16(a, b, acc[q]);
    }
  }

#pragma unroll
  for (int v = 0; v < 8; ++v) {
    int bm = m0 + half8 + v;
    size_t idx = (size_t)bm * HID + h0 + nc;
    float ig = sigmoidf_(acc[0][v]);
    float fg = sigmoidf_(acc[1][v]);
    float gg = tanhf(acc[2][v]);
    float og = sigmoidf_(acc[3][v]);
    float cn = fg * c[idx] + ig * gg;
    c[idx] = cn;
    hnext[idx] = (__bf16)(og * tanhf(cn));
  }
}

// ---------------------------------------------------------------------------
// One spike iteration (two spike phases), fully fused:
//   s1 = memin > 0.5 ; m1 = memin - s1 + s1@Wfc^T + bfc ; s2 = m1 > 0.5
//   memout = m1 - s2 ; out[2t]=s1, out[2t+1]=s2
// s1 is materialized only inside the bf16 A-fragment (exact: 1.0/0.0).
// ---------------------------------------------------------------------------
__global__ __launch_bounds__(256)
void spike_step_kernel(const float* __restrict__ memin,
                       float* __restrict__ memout,
                       const __bf16* __restrict__ Wfc,  // [H, H] bf16
                       const float* __restrict__ bfc,   // [H]
                       float* __restrict__ out, int t) {
  int lane = threadIdx.x;
  int tile = blockIdx.x * blockDim.y + threadIdx.y;   // 16 x 32 = 512 tiles
  int mt = tile / (HID / 16), nt = tile % (HID / 16);
  int m0 = mt * 16, n0 = nt * 16;
  int nc = n0 + (lane & 15);
  int half8 = (lane >> 4) << 3;
  float bv = bfc[nc];
  v8f acc;
#pragma unroll
  for (int v = 0; v < 8; ++v) acc[v] = bv;
  for (int k0 = 0; k0 < HID; k0 += 32) {
    v16bf a = load_spike_a(memin, m0, k0, lane);
    v16bf b = load_b_frag(Wfc, HID, n0, k0, lane);
    acc = wmma_bf16(a, b, acc);
  }
  float* o1 = out + (size_t)(2 * t) * BATCH * HID;
  float* o2 = o1 + (size_t)BATCH * HID;
#pragma unroll
  for (int v = 0; v < 8; ++v) {
    int bm = m0 + half8 + v;
    size_t idx = (size_t)bm * HID + nc;
    float mo = memin[idx];
    float s1 = (mo > 0.5f) ? 1.0f : 0.0f;
    float m1 = mo - s1 + acc[v];            // acc includes bias
    float s2 = (m1 > 0.5f) ? 1.0f : 0.0f;
    o1[idx] = s1;
    o2[idx] = s2;
    memout[idx] = m1 - s2;
  }
}

// ---------------------------------------------------------------------------
// Small utility kernels
// ---------------------------------------------------------------------------
__global__ void cvt_f32_bf16_kernel(const float* __restrict__ in,
                                    __bf16* __restrict__ o, int n) {
  int i = blockIdx.x * blockDim.x + threadIdx.x;
  if (i < n) o[i] = (__bf16)in[i];
}

__global__ void add_bias_kernel(const float* __restrict__ a,
                                const float* __restrict__ b,
                                float* __restrict__ o, int n) {
  int i = blockIdx.x * blockDim.x + threadIdx.x;
  if (i < n) o[i] = a[i] + b[i];
}

__global__ void zero_u32_kernel(unsigned int* __restrict__ p, int n) {
  int i = blockIdx.x * blockDim.x + threadIdx.x;
  if (i < n) p[i] = 0u;
}

// ---------------------------------------------------------------------------
extern "C" void kernel_launch(void* const* d_in, const int* in_sizes, int n_in,
                              void* d_out, int out_size, void* d_ws, size_t ws_size,
                              hipStream_t stream) {
  const float* x   = (const float*)d_in[0];   // [256, 64, 512]
  const float* Wih = (const float*)d_in[1];   // [2048, 512]
  const float* Whh = (const float*)d_in[2];   // [2048, 512]
  const float* bih = (const float*)d_in[3];   // [2048]
  const float* bhh = (const float*)d_in[4];   // [2048]
  const float* Wfc = (const float*)d_in[5];   // [512, 512]
  const float* bfc = (const float*)d_in[6];   // [512]
  float* out = (float*)d_out;                 // [200, 256, 512]

  const size_t XE = (size_t)BATCH * TSEQ * HID;   // 8388608

  char* ws = (char*)d_ws;
  size_t off = 0;
  auto take = [&](size_t bytes) -> char* {
    char* p = ws + off;
    off = (off + bytes + 255) & ~(size_t)255;
    return p;
  };
  __bf16* xb   = (__bf16*)take(XE * 2);
  __bf16* wihb = (__bf16*)take((size_t)GATES * HID * 2);
  __bf16* whhb = (__bf16*)take((size_t)GATES * HID * 2);
  __bf16* wfcb = (__bf16*)take((size_t)HID * HID * 2);
  float*  bias = (float*) take((size_t)GATES * 4);
  __bf16* hbuf[2];
  hbuf[0] = (__bf16*)take((size_t)BATCH * HID * 2);
  hbuf[1] = (__bf16*)take((size_t)BATCH * HID * 2);
  float* cbuf = (float*)take((size_t)BATCH * HID * 4);
  float* mbuf[2];
  mbuf[0] = (float*)take((size_t)BATCH * HID * 4);
  mbuf[1] = (float*)take((size_t)BATCH * HID * 4);
  float* G = (float*)take((size_t)BATCH * TSEQ * GATES * 4);   // 128 MB
  const bool use_g = (off <= ws_size);

  const dim3 wblk(32, 8);   // 8 wave32 tiles per block

  // weight / input conversion + bias fold (parallel, off the critical chain)
  cvt_f32_bf16_kernel<<<(int)((XE + 255) / 256), 256, 0, stream>>>(x, xb, (int)XE);
  cvt_f32_bf16_kernel<<<(GATES * HID + 255) / 256, 256, 0, stream>>>(Wih, wihb, GATES * HID);
  cvt_f32_bf16_kernel<<<(GATES * HID + 255) / 256, 256, 0, stream>>>(Whh, whhb, GATES * HID);
  cvt_f32_bf16_kernel<<<(HID * HID + 255) / 256, 256, 0, stream>>>(Wfc, wfcb, HID * HID);
  add_bias_kernel<<<(GATES + 255) / 256, 256, 0, stream>>>(bih, bhh, bias, GATES);
  zero_u32_kernel<<<(BATCH * HID / 2 + 255) / 256, 256, 0, stream>>>(
      (unsigned int*)hbuf[0], BATCH * HID / 2);
  zero_u32_kernel<<<(BATCH * HID + 255) / 256, 256, 0, stream>>>(
      (unsigned int*)cbuf, BATCH * HID);

  // G = x @ W_ih^T + bias : (B*T/16)*(4H/16) = 1024*128 tiles / 8 per block
  if (use_g) {
    gemm_bias_kernel<<<(BATCH * TSEQ / 16) * (GATES / 16) / 8, wblk, 0, stream>>>(
        xb, HID, wihb, HID, bias, G, GATES, GATES / 16, HID);
  }

  // 64 sequential LSTM steps, h double-buffered; h_last lands in hbuf[0]
  for (int t = 0; t < TSEQ; ++t) {
    lstm_step_kernel<<<(BATCH / 16) * (HID / 16) / 8, wblk, 0, stream>>>(
        use_g ? G : nullptr, xb, wihb, whhb, bias,
        hbuf[t & 1], hbuf[(t + 1) & 1], cbuf, t);
  }

  // mem0 = h_last @ W_fc^T + b_fc
  gemm_bias_kernel<<<(BATCH / 16) * (HID / 16) / 8, wblk, 0, stream>>>(
      hbuf[0], HID, wfcb, HID, bfc, mbuf[0], HID, HID / 16, HID);

  // 100 sequential fused spike iterations, mem double-buffered
  for (int t = 0; t < NSPK; ++t) {
    spike_step_kernel<<<(BATCH / 16) * (HID / 16) / 8, wblk, 0, stream>>>(
        mbuf[t & 1], mbuf[(t + 1) & 1], wfcb, bfc, out, t);
  }
}
